// SelfAttentionAggregator_9620726743346
// MI455X (gfx1250) — matrix operand order repaired and checked by reading, not compiled, not run on hardware
//
#include <hip/hip_runtime.h>
#include <hip/hip_bf16.h>

// ---------------------------------------------------------------------------
// CDNA5 (gfx1250) Nystrom-attention aggregator.
// - Large GEMMs: v_wmma_f32_16x16x32_bf16 (f32 accumulate)
// - Newton-Schulz pinv chain: v_wmma_f32_16x16x4_f32 (stays in f32)
// - Feature GEMM (K=1024) uses async global->LDS staging (ASYNCcnt path,
//   double buffered) with a 64x64 C tile per 4-wave workgroup.
// ---------------------------------------------------------------------------

typedef __attribute__((ext_vector_type(16))) __bf16 bf16x16;
typedef __attribute__((ext_vector_type(8)))  float  v8f;
typedef __attribute__((ext_vector_type(2)))  float  v2f;

#define DEV static __device__ __forceinline__

#define WMMA_BF16(a, b, c) \
  __builtin_amdgcn_wmma_f32_16x16x32_bf16(false, (a), false, (b), (short)0, (c), false, false)
#define WMMA_F32(a, b, c) \
  __builtin_amdgcn_wmma_f32_16x16x4_f32(false, (a), false, (b), (short)0, (c), false, false)

// ---- WMMA operand loaders (f32 per ISA 7.12.2 layouts; bf16 contiguous k, wave32) ---

// bf16 A-matrix 16x32: lanes 0-15 hold row m, K=0..15; lanes 16-31 K=16..31
DEV bf16x16 load_a16(const __bf16* A, int lda, int m0, int k0, int lane) {
  int m  = lane & 15;
  int ko = (lane < 16) ? 0 : 16;
  const __bf16* p = A + (long)(m0 + m) * lda + k0 + ko;
  bf16x16 r;
#pragma unroll
  for (int i = 0; i < 8; ++i) r[i] = p[i];
#pragma unroll
  for (int i = 0; i < 8; ++i) r[8 + i] = p[8 + i];
  return r;
}

// bf16 B-matrix 32x16 from B^T storage (Bt[n][k], K contiguous per column lane)
DEV bf16x16 load_bt16(const __bf16* Bt, int ldbt, int n0, int k0, int lane) {
  int n  = lane & 15;
  int ko = (lane < 16) ? 0 : 16;
  const __bf16* p = Bt + (long)(n0 + n) * ldbt + k0 + ko;
  bf16x16 r;
#pragma unroll
  for (int i = 0; i < 16; ++i) r[i] = p[i];
  return r;
}

// f32 A-matrix 16x4: lanes 0-15 K=0,1; lanes 16-31 K=2,3
DEV v2f load_a32(const float* A, int lda, int m0, int k0, int lane) {
  int m  = lane & 15;
  int ko = (lane < 16) ? 0 : 2;
  const float* p = A + (long)(m0 + m) * lda + k0 + ko;
  v2f r; r.x = p[0]; r.y = p[1];
  return r;
}

// f32 B-matrix 4x16 from row-major B[k][n] (coalesced across lanes)
DEV v2f load_b32(const float* B, int ldb, int k0, int n0, int lane) {
  int n  = lane & 15;
  int ko = (lane < 16) ? 0 : 2;
  v2f r;
  r.x = B[(long)(k0 + ko    ) * ldb + n0 + n];
  r.y = B[(long)(k0 + ko + 1) * ldb + n0 + n];
  return r;
}

// ---- CDNA5 async global -> LDS (ASYNCcnt) ---------------------------------

DEV void async_copy16(unsigned lds_off, const void* gaddr) {
  asm volatile("global_load_async_to_lds_b128 %0, %1, off"
               :: "v"(lds_off), "v"(gaddr) : "memory");
}
DEV void wait_async0() {
  asm volatile("s_wait_asynccnt 0x0" ::: "memory");
}

// ---- conversion helpers ---------------------------------------------------

__global__ void k_cvt(const float* src, __bf16* dst, long n) {
  long i = (long)blockIdx.x * 256 + threadIdx.x;
  if (i < n) dst[i] = (__bf16)src[i];
}

// src is KxN row-major, dst is NxK row-major (B^T for contiguous WMMA B loads)
__global__ void k_cvt_tr(const float* src, __bf16* dst, int K, int N) {
  long i = (long)blockIdx.x * 256 + threadIdx.x;
  if (i < (long)K * N) {
    int kk = (int)(i / N), nn = (int)(i % N);
    dst[(long)nn * K + kk] = (__bf16)src[i];
  }
}

__global__ void k_init(unsigned* scale_bits) {
  if (threadIdx.x == 0) *scale_bits = 0u;
}

// ---- h = relu(x @ W_ft + b_ft)   M=8192 K=1024 N=256 ----------------------
// Workgroup = 4 waves, 64x64 C tile; A/B tiles double-buffered in LDS via
// async global->LDS copies.

__global__ void k_gemm_h(const __bf16* A, const __bf16* Bt, const float* bias,
                         float* Hf, __bf16* Hb) {
  __shared__ __bf16 sA[2][64 * 32];   // 64 rows x 32 k, 4 KB each
  __shared__ __bf16 sB[2][64 * 32];   // 64 cols x 32 k (B^T rows), 4 KB each
  int tid = threadIdx.x, lane = tid & 31, w = tid >> 5;
  int n0 = blockIdx.x * 64, m0 = blockIdx.y * 64;

  // stage tile kt (K offset kt*32) into buffer buf
  auto stage = [&](int buf, int k0) {
    unsigned la = (unsigned)(size_t)&sA[buf][0];
    unsigned lb = (unsigned)(size_t)&sB[buf][0];
#pragma unroll
    for (int c = 0; c < 2; ++c) {
      int chunk = tid + c * 128;            // 0..255 -> 16B chunks of 4 KB tile
      int row = chunk >> 2;                 // 0..63
      int off = (chunk & 3) * 16;           // byte offset within 64 B row
      const char* ga = (const char*)(A  + (long)(m0 + row) * 1024 + k0) + off;
      const char* gb = (const char*)(Bt + (long)(n0 + row) * 1024 + k0) + off;
      async_copy16(la + (unsigned)(row * 64 + off), ga);
      async_copy16(lb + (unsigned)(row * 64 + off), gb);
    }
  };

  v8f acc0 = {}, acc1 = {}, acc2 = {}, acc3 = {};
  stage(0, 0);
  for (int kt = 0; kt < 32; ++kt) {
    wait_async0();          // my copies done; barrier makes everyone's visible
    __syncthreads();
    if (kt + 1 < 32) stage((kt + 1) & 1, (kt + 1) * 32);
    const __bf16* la = &sA[kt & 1][0];
    const __bf16* lb = &sB[kt & 1][0];
    bf16x16 a = load_a16(la, 32, w * 16, 0, lane);
    bf16x16 b0 = load_bt16(lb, 32,  0, 0, lane);
    bf16x16 b1 = load_bt16(lb, 32, 16, 0, lane);
    bf16x16 b2 = load_bt16(lb, 32, 32, 0, lane);
    bf16x16 b3 = load_bt16(lb, 32, 48, 0, lane);
    acc0 = WMMA_BF16(a, b0, acc0);
    acc1 = WMMA_BF16(a, b1, acc1);
    acc2 = WMMA_BF16(a, b2, acc2);
    acc3 = WMMA_BF16(a, b3, acc3);
    __syncthreads();
  }

  int n = lane & 15, mo = (lane < 16) ? 0 : 8;
#pragma unroll
  for (int r = 0; r < 8; ++r) {
    int row = m0 + w * 16 + mo + r;
    v8f* accs[4] = {&acc0, &acc1, &acc2, &acc3};
#pragma unroll
    for (int nt = 0; nt < 4; ++nt) {
      int col = n0 + nt * 16 + n;
      float v = fmaxf((*accs[nt])[r] + bias[col], 0.f);
      long idx = (long)row * 256 + col;
      Hf[idx] = v;
      Hb[idx] = (__bf16)v;
    }
  }
}

// ---- qkv = h @ W_qkv, scatter to (head, n, d); q scaled by 64^-0.5 --------

__global__ void k_gemm_qkv(const __bf16* A, const __bf16* Bt,
                           __bf16* q_s, __bf16* k_b, __bf16* v_b) {
  int lane = threadIdx.x;
  int n0 = blockIdx.x * 16, m0 = blockIdx.y * 16;
  v8f c = {};
  for (int k0 = 0; k0 < 256; k0 += 32) {
    bf16x16 a = load_a16(A, 256, m0, k0, lane);
    bf16x16 b = load_bt16(Bt, 256, n0, k0, lane);
    c = WMMA_BF16(a, b, c);
  }
  int n = lane & 15, mo = (lane < 16) ? 0 : 8;
  int col = n0 + n;
  int third = col >> 9, hc = col & 511, head = hc >> 6, d = hc & 63;
#pragma unroll
  for (int r = 0; r < 8; ++r) {
    int row = m0 + mo + r;
    long dst = ((long)head * 8192 + row) * 64 + d;
    float v = c[r];
    if (third == 0)      q_s[dst] = (__bf16)(v * 0.125f);
    else if (third == 1) k_b[dst] = (__bf16)v;
    else                 v_b[dst] = (__bf16)v;
  }
}

// ---- transpose v to (head, d, n) so the K=8192 GEMM gets contiguous B -----

__global__ void k_tr_v(const __bf16* v_b, __bf16* vT) {
  long i = (long)blockIdx.x * 256 + threadIdx.x;   // 8*8192*64
  int h = (int)(i >> 19), nn = (int)((i >> 6) & 8191), d = (int)(i & 63);
  vT[(long)h * 524288 + (long)d * 8192 + nn] = v_b[i];
}

// ---- landmark means: (8,8192,64) -> (8,256,64), mean over 32 tokens -------

__global__ void k_land(const __bf16* src, __bf16* dst) {
  int i = blockIdx.x * 256 + threadIdx.x;   // 8*256*64 = 131072
  int h = i >> 14, m = (i >> 6) & 255, d = i & 63;
  const __bf16* p = src + ((long)h * 8192 + m * 32) * 64 + d;
  float s = 0.f;
#pragma unroll
  for (int t = 0; t < 32; ++t) s += (float)p[t * 64];
  dst[i] = (__bf16)(s * (1.f / 32.f));
}

// ---- generic batched bf16 GEMM, B^T-layout B, optional f32/bf16 out -------

__global__ void k_gemm_bf16_bt(const __bf16* A, const __bf16* Bt,
                               float* Df, __bf16* Db,
                               int K, int lda, int ldbt, int ldd,
                               long sA, long sBt, long sD) {
  int lane = threadIdx.x;
  int n0 = blockIdx.x * 16, m0 = blockIdx.y * 16, bz = blockIdx.z;
  const __bf16* Ab = A  + (long)bz * sA;
  const __bf16* Bb = Bt + (long)bz * sBt;
  v8f c = {};
  for (int k0 = 0; k0 < K; k0 += 32) {
    bf16x16 a = load_a16(Ab, lda, m0, k0, lane);
    bf16x16 b = load_bt16(Bb, ldbt, n0, k0, lane);
    c = WMMA_BF16(a, b, c);
  }
  int n = lane & 15, mo = (lane < 16) ? 0 : 8;
#pragma unroll
  for (int r = 0; r < 8; ++r) {
    long idx = (long)bz * sD + (long)(m0 + mo + r) * ldd + n0 + n;
    if (Df) Df[idx] = c[r];
    if (Db) Db[idx] = (__bf16)c[r];
  }
}

// ---- row softmax, in place (each element touched only by its own thread) --

__global__ void k_softmax_f32(float* X, int C) {
  float* p = X + (long)blockIdx.x * C;
  __shared__ float red[256];
  int tid = threadIdx.x;
  float mx = -1e30f;
  for (int c = tid; c < C; c += 256) mx = fmaxf(mx, p[c]);
  red[tid] = mx; __syncthreads();
  for (int s = 128; s > 0; s >>= 1) { if (tid < s) red[tid] = fmaxf(red[tid], red[tid + s]); __syncthreads(); }
  mx = red[0]; __syncthreads();
  float sum = 0.f;
  for (int c = tid; c < C; c += 256) sum += expf(p[c] - mx);
  red[tid] = sum; __syncthreads();
  for (int s = 128; s > 0; s >>= 1) { if (tid < s) red[tid] += red[tid + s]; __syncthreads(); }
  float inv = 1.f / red[0];
  for (int c = tid; c < C; c += 256) p[c] = expf(p[c] - mx) * inv;
}

__global__ void k_softmax_bf16(__bf16* X, int C) {
  __bf16* p = X + (long)blockIdx.x * C;
  __shared__ float red[256];
  int tid = threadIdx.x;
  float mx = -1e30f;
  for (int c = tid; c < C; c += 256) mx = fmaxf(mx, (float)p[c]);
  red[tid] = mx; __syncthreads();
  for (int s = 128; s > 0; s >>= 1) { if (tid < s) red[tid] = fmaxf(red[tid], red[tid + s]); __syncthreads(); }
  mx = red[0]; __syncthreads();
  float sum = 0.f;
  for (int c = tid; c < C; c += 256) sum += expf((float)p[c] - mx);
  red[tid] = sum; __syncthreads();
  for (int s = 128; s > 0; s >>= 1) { if (tid < s) red[tid] += red[tid + s]; __syncthreads(); }
  float inv = 1.f / red[0];
  for (int c = tid; c < C; c += 256) p[c] = (__bf16)(expf((float)p[c] - mx) * inv);
}

// ---- pinv scale: rows of softmax sum to 1, so scale == max column sum -----

__global__ void k_colscale(const float* attn2, unsigned* scale_bits) {
  int h = blockIdx.x, c = threadIdx.x;
  const float* p = attn2 + (long)h * 65536;
  float s = 0.f;
  for (int m = 0; m < 256; ++m) s += p[m * 256 + c];
  __shared__ float red[256];
  red[c] = s; __syncthreads();
  for (int st = 128; st > 0; st >>= 1) { if (c < st) red[c] = fmaxf(red[c], red[c + st]); __syncthreads(); }
  if (c == 0) atomicMax(scale_bits, __float_as_uint(red[0]));  // values >= 0
}

__global__ void k_zinit(const float* attn2, const unsigned* scale_bits, float* z) {
  long i = (long)blockIdx.x * 256 + threadIdx.x;   // 8*256*256
  float inv = 1.f / __uint_as_float(*scale_bits);
  int h = (int)(i >> 16), r = (int)((i >> 8) & 255), c = (int)(i & 255);
  z[i] = attn2[((long)h << 16) + (long)c * 256 + r] * inv;   // transpose / scale
}

// ---- batched f32 matmul with diag epilogue: D = s*acc + d*I (x2 outputs) --

__global__ void k_mm_f32(const float* A, const float* B,
                         float* D1, float s1, float d1,
                         float* D2, float s2, float d2,
                         int K, int lda, int ldb, int ldd,
                         long sA, long sB, long sD) {
  int lane = threadIdx.x;
  int n0 = blockIdx.x * 16, m0 = blockIdx.y * 16, bz = blockIdx.z;
  const float* Ab = A + (long)bz * sA;
  const float* Bb = B + (long)bz * sB;
  v8f c = {};
  for (int k0 = 0; k0 < K; k0 += 4) {
    v2f a = load_a32(Ab, lda, m0, k0, lane);
    v2f b = load_b32(Bb, ldb, k0, n0, lane);
    c = WMMA_F32(a, b, c);
  }
  int n = lane & 15, mo = (lane < 16) ? 0 : 8;
#pragma unroll
  for (int r = 0; r < 8; ++r) {
    int row = m0 + mo + r, col = n0 + n;
    float dg = (row == col) ? 1.f : 0.f;
    long idx = (long)bz * sD + (long)row * ldd + col;
    D1[idx] = s1 * c[r] + d1 * dg;
    if (D2) D2[idx] = s2 * c[r] + d2 * dg;
  }
}

// ---- ov = attn3 @ v   (8: 256x8192 @ 8192x64), B from v^T -----------------

__global__ void k_av(const __bf16* attn3, const __bf16* vT, float* ov) {
  int lane = threadIdx.x;
  int n0 = blockIdx.x * 16, m0 = blockIdx.y * 16, h = blockIdx.z;
  const __bf16* Ab = attn3 + (long)h * 256 * 8192;
  const __bf16* Bb = vT    + (long)h * 524288;       // (64, 8192) = B^T
  v8f c = {};
  for (int k0 = 0; k0 < 8192; k0 += 32) {
    bf16x16 a = load_a16(Ab, 8192, m0, k0, lane);
    bf16x16 b = load_bt16(Bb, 8192, n0, k0, lane);
    c = WMMA_BF16(a, b, c);
  }
  int n = lane & 15, mo = (lane < 16) ? 0 : 8;
#pragma unroll
  for (int r = 0; r < 8; ++r)
    ov[((long)h * 256 + m0 + mo + r) * 64 + n0 + n] = c[r];
}

// ---- wT = (pinv @ ov)^T as bf16, per head --------------------------------

__global__ void k_w(const float* z, const float* ov, __bf16* wT) {
  int lane = threadIdx.x;
  int n0 = blockIdx.x * 16, m0 = blockIdx.y * 16, h = blockIdx.z;
  const float* Ab = z  + (long)h * 65536;
  const float* Bb = ov + (long)h * 256 * 64;
  v8f c = {};
  for (int k0 = 0; k0 < 256; k0 += 4) {
    v2f a = load_a32(Ab, 256, m0, k0, lane);
    v2f b = load_b32(Bb, 64, k0, n0, lane);
    c = WMMA_F32(a, b, c);
  }
  int n = lane & 15, mo = (lane < 16) ? 0 : 8;
#pragma unroll
  for (int r = 0; r < 8; ++r)
    wT[(long)h * 64 * 256 + (long)(n0 + n) * 256 + m0 + mo + r] = (__bf16)c[r];
}

// ---- fused: out = softmax(q @ k_l^T) @ w   (attn1 never hits HBM) ---------
// block = 64 threads (2 waves), 32 q-rows per block

__global__ void k_out(const __bf16* q_s, const __bf16* k_l, const __bf16* wT,
                      float* out_attn) {
  __shared__ float  s_s[32][256];   // sim1 tile (32 KB)
  __shared__ __bf16 s_a[32][256];   // attn1 tile bf16 (16 KB)
  int tid = threadIdx.x, lane = tid & 31, w = tid >> 5;
  int head = blockIdx.y;
  int gm0 = blockIdx.x * 32 + w * 16;
  const __bf16* Aq  = q_s + (long)head * 8192 * 64;
  const __bf16* Bkl = k_l + (long)head * 256 * 64;
  int n = lane & 15, mo = (lane < 16) ? 0 : 8;

  for (int nt = 0; nt < 16; ++nt) {
    v8f c = {};
    for (int k0 = 0; k0 < 64; k0 += 32) {
      bf16x16 a = load_a16(Aq, 64, gm0, k0, lane);
      bf16x16 b = load_bt16(Bkl, 64, nt * 16, k0, lane);
      c = WMMA_BF16(a, b, c);
    }
#pragma unroll
    for (int r = 0; r < 8; ++r) s_s[w * 16 + mo + r][nt * 16 + n] = c[r];
  }
  __syncthreads();

  if (tid < 32) {                        // one thread per row: 256-wide softmax
    float mx = -1e30f;
    for (int c2 = 0; c2 < 256; ++c2) mx = fmaxf(mx, s_s[tid][c2]);
    float sum = 0.f;
    for (int c2 = 0; c2 < 256; ++c2) { float e = expf(s_s[tid][c2] - mx); s_s[tid][c2] = e; sum += e; }
    float inv = 1.f / sum;
    for (int c2 = 0; c2 < 256; ++c2) s_a[tid][c2] = (__bf16)(s_s[tid][c2] * inv);
  }
  __syncthreads();

  const __bf16* Bw = wT + (long)head * 64 * 256;
  for (int nt = 0; nt < 4; ++nt) {
    v8f c = {};
    for (int k0 = 0; k0 < 256; k0 += 32) {
      bf16x16 a = load_a16(&s_a[0][0], 256, w * 16, k0, lane);
      bf16x16 b = load_bt16(Bw, 256, nt * 16, k0, lane);
      c = WMMA_BF16(a, b, c);
    }
#pragma unroll
    for (int r = 0; r < 8; ++r) {
      long g = (long)blockIdx.x * 32 + w * 16 + mo + r;
      out_attn[((long)head * 8192 + g) * 64 + nt * 16 + n] = c[r];
    }
  }
}

// ---- depthwise conv-33 residual + head-concat to (n, 512) bf16 ------------

__global__ void k_conv_cat(const float* out_attn, const __bf16* v_b, const float* cw,
                           __bf16* out_cat) {
  long i = (long)blockIdx.x * 256 + threadIdx.x;   // 8192*512
  int nn = (int)(i >> 9), cc = (int)(i & 511), h = cc >> 6, d = cc & 63;
  float acc = out_attn[((long)h * 8192 + nn) * 64 + d];
#pragma unroll
  for (int j = 0; j < 33; ++j) {
    int t = nn + j - 16;
    if (t >= 0 && t < 8192)
      acc += cw[h * 33 + j] * (float)v_b[((long)h * 8192 + t) * 64 + d];
  }
  out_cat[i] = (__bf16)acc;
}

// ---- fused: af = cat @ W_out + b_out; scores = relu(af @ W_score + b) -----

__global__ void k_proj_score(const __bf16* A, const __bf16* Bt,
                             const float* b_out, const float* W_score,
                             const float* b_score, float* scores) {
  __shared__ float af[32][256];
  int tid = threadIdx.x, lane = tid & 31, w = tid >> 5;
  int gm0 = blockIdx.x * 32 + w * 16;
  int n = lane & 15, mo = (lane < 16) ? 0 : 8;
  for (int nt = 0; nt < 16; ++nt) {
    v8f c = {};
    for (int k0 = 0; k0 < 512; k0 += 32) {
      bf16x16 a = load_a16(A, 512, gm0, k0, lane);
      bf16x16 b = load_bt16(Bt, 512, nt * 16, k0, lane);
      c = WMMA_BF16(a, b, c);
    }
#pragma unroll
    for (int r = 0; r < 8; ++r)
      af[w * 16 + mo + r][nt * 16 + n] = c[r] + b_out[nt * 16 + n];
  }
  __syncthreads();
  if (tid < 32) {
    float s = b_score[0];
    for (int c2 = 0; c2 < 256; ++c2) s += af[tid][c2] * W_score[c2];
    scores[blockIdx.x * 32 + tid] = fmaxf(s, 0.f);
  }
}

// ---- final softmax over N and aggregation ---------------------------------

__global__ void k_softmax_final(const float* scores, float* weights) {
  __shared__ float red[1024];
  int tid = threadIdx.x;
  float mx = -1e30f;
  for (int i = tid; i < 8192; i += 1024) mx = fmaxf(mx, scores[i]);
  red[tid] = mx; __syncthreads();
  for (int s = 512; s > 0; s >>= 1) { if (tid < s) red[tid] = fmaxf(red[tid], red[tid + s]); __syncthreads(); }
  mx = red[0]; __syncthreads();
  float sum = 0.f;
  for (int i = tid; i < 8192; i += 1024) sum += expf(scores[i] - mx);
  red[tid] = sum; __syncthreads();
  for (int s = 512; s > 0; s >>= 1) { if (tid < s) red[tid] += red[tid + s]; __syncthreads(); }
  float inv = 1.f / red[0];
  for (int i = tid; i < 8192; i += 1024) weights[i] = expf(scores[i] - mx) * inv;
}

__global__ void k_agg(const float* weights, const float* h_f32, float* aggregated) {
  int col = blockIdx.x, tid = threadIdx.x;
  float s = 0.f;
  for (int nn = tid; nn < 8192; nn += 256) s += weights[nn] * h_f32[(long)nn * 256 + col];
  __shared__ float red[256];
  red[tid] = s; __syncthreads();
  for (int st = 128; st > 0; st >>= 1) { if (tid < st) red[tid] += red[tid + st]; __syncthreads(); }
  if (tid == 0) aggregated[col] = red[0];
}

// ---------------------------------------------------------------------------

extern "C" void kernel_launch(void* const* d_in, const int* in_sizes, int n_in,
                              void* d_out, int out_size, void* d_ws, size_t ws_size,
                              hipStream_t stream) {
  (void)in_sizes; (void)n_in; (void)out_size; (void)ws_size;
  const float* x       = (const float*)d_in[0];
  const float* W_ft    = (const float*)d_in[1];
  const float* b_ft    = (const float*)d_in[2];
  const float* W_qkv   = (const float*)d_in[3];
  const float* W_out   = (const float*)d_in[4];
  const float* b_out   = (const float*)d_in[5];
  const float* conv_w  = (const float*)d_in[6];
  const float* W_score = (const float*)d_in[7];
  const float* b_score = (const float*)d_in[8];
  float* out = (float*)d_out;                 // [0..255]=aggregated, [256..8447]=weights

  char* p = (char*)d_ws;
  auto alloc = [&](size_t bytes) -> void* {
    void* r = (void*)p; p += (bytes + 255) & ~(size_t)255; return r;
  };
  __bf16* x_b    = (__bf16*)alloc(8192L * 1024 * 2);
  __bf16* WftT   = (__bf16*)alloc(256L * 1024 * 2);
  __bf16* WqkvT  = (__bf16*)alloc(1536L * 256 * 2);
  __bf16* WoutT  = (__bf16*)alloc(256L * 512 * 2);
  float*  h_f    = (float*)alloc(8192L * 256 * 4);
  __bf16* h_b    = (__bf16*)alloc(8192L * 256 * 2);
  __bf16* q_s    = (__bf16*)alloc(8L * 8192 * 64 * 2);
  __bf16* k_b    = (__bf16*)alloc(8L * 8192 * 64 * 2);
  __bf16* v_b    = (__bf16*)alloc(8L * 8192 * 64 * 2);
  __bf16* vT     = (__bf16*)alloc(8L * 64 * 8192 * 2);
  __bf16* q_l    = (__bf16*)alloc(8L * 256 * 64 * 2);
  __bf16* k_l    = (__bf16*)alloc(8L * 256 * 64 * 2);
  float*  attn2  = (float*)alloc(8L * 65536 * 4);
  float*  zb0    = (float*)alloc(8L * 65536 * 4);
  float*  zb1    = (float*)alloc(8L * 65536 * 4);
  float*  XZ     = (float*)alloc(8L * 65536 * 4);
  float*  T1     = (float*)alloc(8L * 65536 * 4);
  float*  T3     = (float*)alloc(8L * 65536 * 4);
  float*  T5     = (float*)alloc(8L * 65536 * 4);
  __bf16* sim3   = (__bf16*)alloc(8L * 256 * 8192 * 2);
  float*  ov     = (float*)alloc(8L * 256 * 64 * 4);
  __bf16* wT     = (__bf16*)alloc(8L * 64 * 256 * 2);
  float*  o_att  = (float*)alloc(8L * 8192 * 64 * 4);
  __bf16* o_cat  = (__bf16*)alloc(8192L * 512 * 2);
  float*  scores = (float*)alloc(8192 * 4);
  unsigned* scale_bits = (unsigned*)alloc(256);

  k_init<<<1, 32, 0, stream>>>(scale_bits);

  // convert x + pre-transpose weights to bf16
  k_cvt<<<32768, 256, 0, stream>>>(x, x_b, 8192L * 1024);
  k_cvt_tr<<<1024, 256, 0, stream>>>(W_ft, WftT, 1024, 256);
  k_cvt_tr<<<1536, 256, 0, stream>>>(W_qkv, WqkvT, 256, 1536);
  k_cvt_tr<<<512, 256, 0, stream>>>(W_out, WoutT, 512, 256);

  // h = relu(x @ W_ft + b_ft)  (async-LDS double-buffered WMMA GEMM)
  k_gemm_h<<<dim3(4, 128), 128, 0, stream>>>(x_b, WftT, b_ft, h_f, h_b);

  // qkv + head scatter (q scaled)
  k_gemm_qkv<<<dim3(96, 512), 32, 0, stream>>>(h_b, WqkvT, q_s, k_b, v_b);
  k_tr_v<<<16384, 256, 0, stream>>>(v_b, vT);

  // landmark means
  k_land<<<512, 256, 0, stream>>>(q_s, q_l);
  k_land<<<512, 256, 0, stream>>>(k_b, k_l);

  // attn2 = softmax(q_l @ k_l^T)
  k_gemm_bf16_bt<<<dim3(16, 16, 8), 32, 0, stream>>>(
      q_l, k_l, attn2, (__bf16*)nullptr, 64, 64, 64, 256, 16384, 16384, 65536);
  k_softmax_f32<<<2048, 256, 0, stream>>>(attn2, 256);

  // pinv init: z = attn2^T / (max col sum)   (row sums of softmax are 1)
  k_colscale<<<8, 256, 0, stream>>>(attn2, scale_bits);
  k_zinit<<<2048, 256, 0, stream>>>(attn2, scale_bits, zb0);

  // 6 Newton-Schulz iterations, all v_wmma_f32_16x16x4_f32
  float* zc = zb0; float* zn = zb1;
  for (int it = 0; it < 6; ++it) {
    dim3 g(16, 16, 8);
    k_mm_f32<<<g, 32, 0, stream>>>(attn2, zc, XZ, 1.f, 0.f, T1, -1.f, 7.f,
                                   256, 256, 256, 256, 65536, 65536, 65536);
    k_mm_f32<<<g, 32, 0, stream>>>(XZ, T1, T3, -1.f, 15.f, (float*)nullptr, 0.f, 0.f,
                                   256, 256, 256, 256, 65536, 65536, 65536);
    k_mm_f32<<<g, 32, 0, stream>>>(XZ, T3, T5, -1.f, 13.f, (float*)nullptr, 0.f, 0.f,
                                   256, 256, 256, 256, 65536, 65536, 65536);
    k_mm_f32<<<g, 32, 0, stream>>>(zc, T5, zn, 0.25f, 0.f, (float*)nullptr, 0.f, 0.f,
                                   256, 256, 256, 256, 65536, 65536, 65536);
    float* t = zc; zc = zn; zn = t;
  }

  // attn3 = softmax(q_l @ k^T) in place (bf16), then ov = attn3 @ v
  k_gemm_bf16_bt<<<dim3(512, 16, 8), 32, 0, stream>>>(
      q_l, k_b, (float*)nullptr, sim3, 64, 64, 64, 8192, 16384, 524288, 2097152);
  k_softmax_bf16<<<2048, 256, 0, stream>>>(sim3, 8192);
  k_av<<<dim3(4, 16, 8), 32, 0, stream>>>(sim3, vT, ov);

  // wT = (pinv @ ov)^T
  k_w<<<dim3(4, 16, 8), 32, 0, stream>>>(zc, ov, wT);

  // fused sim1 -> softmax -> @ w
  k_out<<<dim3(256, 8), 64, 0, stream>>>(q_s, k_l, wT, o_att);

  // conv residual + concat heads
  k_conv_cat<<<16384, 256, 0, stream>>>(o_att, v_b, conv_w, o_cat);

  // output projection fused with score head
  k_proj_score<<<256, 64, 0, stream>>>(o_cat, WoutT, b_out, W_score, b_score, scores);

  // weights = softmax(scores); aggregated = weights @ h
  k_softmax_final<<<1, 1024, 0, stream>>>(scores, out + 256);
  k_agg<<<256, 256, 0, stream>>>(out + 256, h_f, out);
}
